// NCutLoss3D_8684423872907
// MI455X (gfx1250) — matrix operand order repaired and checked by reading, not compiled, not run on hardware
//
#include <hip/hip_runtime.h>
#include <math.h>

// ---------------- problem constants ----------------
#define DIM    160
#define SY     160
#define SZ     (160*160)
#define NVOL   (160*160*160)             // 4,096,000 elements per batch volume
#define NB     4
#define NTOT   ((size_t)NB * (size_t)NVOL)
#define CHUNK  16384
#define NCHUNK 1000                      // NTOT / CHUNK (exact)
#define CBLOCKS 6400                     // conv blocks per pass: NB*DIM*10

// ---------------- vector types ----------------
typedef __attribute__((ext_vector_type(16))) __bf16         v16bf;
typedef __attribute__((ext_vector_type(8)))  float          v8f;
typedef __attribute__((ext_vector_type(8)))  unsigned short us8;
typedef __attribute__((ext_vector_type(16))) unsigned short us16;
typedef __attribute__((ext_vector_type(4)))  unsigned int   u32x4;
typedef __attribute__((ext_vector_type(8)))  int            i32x8;
typedef __attribute__((ext_vector_type(4)))  int            i32x4;

union Frag { v16bf bf; us16 us; };

__device__ __forceinline__ v8f wmma_bf16(v16bf a, v16bf b, v8f c) {
  // D = A(16x32 bf16) x B(32x16 bf16) + C(16x16 f32)
  return __builtin_amdgcn_wmma_f32_16x16x32_bf16(
      false, a, false, b, (short)0, c, false, false);
}

// 1D Gaussian tap: kern[t] = exp(-(t-4)^2 / (2*sigma1^2)), sigma1=5, t in [0,8]
__device__ __forceinline__ float kern1d(int t) {
  if (t < 0 || t > 8) return 0.0f;
  float d = (float)(t - 4);
  return __expf(-d * d * (1.0f / 50.0f));
}

__device__ __forceinline__ unsigned short bf16_bits(float f) {
  __bf16 h = (__bf16)f;
  return __builtin_bit_cast(unsigned short, h);
}

// ---------------- Tensor Data Mover: 2D tile global->LDS ----------------
// Packs a D# per CDNA5 ISA ch.8 and issues TENSOR_LOAD_TO_LDS (6-arg builtin).
//   data_size_log: 0=1B, 1=2B, 2=4B
//   tile_d0: elements per row (contiguous); tile_d1: rows
//   row_stride: elements between rows in global memory
// Tensor dims are set huge so no OOB clipping occurs (tiles never leave volume).
__device__ __forceinline__ void tdm_load_2d(unsigned lds_addr, const void* gptr,
                                            unsigned data_size_log,
                                            unsigned tile_d0, unsigned tile_d1,
                                            unsigned long long row_stride) {
  unsigned long long ga = (unsigned long long)gptr;
  const unsigned td0 = 0x40000000u, td1 = 0x40000000u;
  u32x4 g0;
  g0[0] = 1u;                                            // count=1, user-mode, no gather
  g0[1] = lds_addr;                                      // LDS byte address
  g0[2] = (unsigned)(ga & 0xFFFFFFFFu);                  // global_addr[31:0]
  g0[3] = (unsigned)((ga >> 32) & 0x01FFFFFFu) | (2u << 30);  // [56:32] | type=2
  i32x8 g1;
  g1[0] = (int)(data_size_log << 16);                    // wg_mask=0, data_size
  g1[1] = (int)((td0 & 0xFFFFu) << 16);                  // atomic_addr=0 | tensor_dim0 lo16
  g1[2] = (int)((td0 >> 16) | ((td1 & 0xFFFFu) << 16));  // tensor_dim0 hi | tensor_dim1 lo
  g1[3] = (int)((td1 >> 16) | (tile_d0 << 16));          // tensor_dim1 hi | tile_dim0
  g1[4] = (int)(tile_d1 & 0xFFFFu);                      // tile_dim1 | tile_dim2=0
  g1[5] = (int)(row_stride & 0xFFFFFFFFull);             // tensor_dim0_stride[31:0]
  g1[6] = (int)((row_stride >> 32) & 0xFFFFull);         // stride[47:32] | dim1_stride lo=0
  g1[7] = 0;
  i32x4 z4 = {0, 0, 0, 0};                               // groups 2/3: 2D tensor
  i32x8 z8 = {0, 0, 0, 0, 0, 0, 0, 0};
  __builtin_amdgcn_tensor_load_to_lds(g0, g1, z4, z4, z8, 0);
}

__device__ __forceinline__ unsigned lds_addr_of(const void* p) {
  return (unsigned)(size_t)p;   // LDS aperture: low 32 bits = workgroup LDS offset
}

// ============ stage 1: per-chunk partial sums of p, in, in*p ============
__global__ void ncut_red1(const float* __restrict__ labels,
                          const float* __restrict__ inputs,
                          float* __restrict__ part) {
  __shared__ float sp[256], si[256], sip[256];
  const int tid = threadIdx.x;
  const size_t base = (size_t)blockIdx.x * CHUNK;
  float a = 0.f, b = 0.f, c = 0.f;
  for (int j = tid; j < CHUNK; j += 256) {
    float L = labels[base + j];
    float I = inputs[base + j];
    a += L; b += I; c += I * L;
  }
  sp[tid] = a; si[tid] = b; sip[tid] = c;
  __syncthreads();
  for (int s = 128; s > 0; s >>= 1) {
    if (tid < s) { sp[tid] += sp[tid+s]; si[tid] += si[tid+s]; sip[tid] += sip[tid+s]; }
    __syncthreads();
  }
  if (tid == 0) {
    part[blockIdx.x*3 + 0] = sp[0];
    part[blockIdx.x*3 + 1] = si[0];
    part[blockIdx.x*3 + 2] = sip[0];
  }
}

// ============ stage 2: per-batch class means ============
__global__ void ncut_red2(const float* __restrict__ part, float* __restrict__ means) {
  __shared__ float s0[256], s1[256], s2[256];
  const int b = blockIdx.x, tid = threadIdx.x;
  float v0 = 0.f, v1 = 0.f, v2 = 0.f;
  if (tid < 250) {
    const float* p = part + ((size_t)b * 250 + tid) * 3;
    v0 = p[0]; v1 = p[1]; v2 = p[2];
  }
  s0[tid] = v0; s1[tid] = v1; s2[tid] = v2;
  __syncthreads();
  for (int s = 128; s > 0; s >>= 1) {
    if (tid < s) { s0[tid] += s0[tid+s]; s1[tid] += s1[tid+s]; s2[tid] += s2[tid+s]; }
    __syncthreads();
  }
  if (tid == 0) {
    const float N = (float)NVOL;
    float sum_p = s0[0], sum_i = s1[0], sum_ip = s2[0];
    means[b*2 + 0] = (sum_ip / N) / (sum_p / N + 1e-5f);
    means[b*2 + 1] = ((sum_i - sum_ip) / N) / ((N - sum_p) / N + 1e-5f);
  }
}

// ============ elementwise field: q==0 -> p*w, q==1 -> w (bf16) ============
__global__ void ncut_weights(const float* __restrict__ labels,
                             const float* __restrict__ inputs,
                             const float* __restrict__ means,
                             __bf16* __restrict__ out, int cls, int q) {
  const size_t stride = (size_t)gridDim.x * blockDim.x;
  for (size_t i = (size_t)blockIdx.x * blockDim.x + threadIdx.x; i < NTOT; i += stride) {
    int b = (int)(i / (size_t)NVOL);
    float L = labels[i], I = inputs[i];
    float p = cls ? (1.0f - L) : L;
    float mean = means[b*2 + cls];
    float d = I - mean; d = d * d;        // diff = (in-mean)^2
    float w = __expf(-d * d);             // exp(-diff^2), sigma2=1
    out[i] = (__bf16)(q ? w : p * w);
  }
}

// ============ pass X: TDM-stage 16y x 160x strip, WMMA per x-tile ============
// block: (b, z, y-tile); 10 waves, wave w -> x-tile X0=16w
__global__ void __launch_bounds__(320) ncut_conv_x(const __bf16* __restrict__ src,
                                                   __bf16* __restrict__ dst) {
  __shared__ __align__(16) unsigned short s_in[16 * DIM];   // [row][x]
  __shared__ __align__(16) unsigned short s_out[16 * DIM];
  const int blk = blockIdx.x;
  const int ty = blk % 10, z = (blk / 10) % DIM, b = blk / 1600;
  const size_t base = (size_t)b * NVOL + (size_t)z * SZ + (size_t)(ty * 16) * SY;

  if (threadIdx.x < 32) {
    tdm_load_2d(lds_addr_of(s_in), src + base, /*2B*/1, DIM, 16, SY);
    __builtin_amdgcn_s_wait_tensorcnt(0);
  }
  __syncthreads();

  const int wave = threadIdx.x >> 5, lane = threadIdx.x & 31;
  const int m = lane & 15, h = lane >> 4, n = m;
  const int X0 = wave * 16;
  const int xa = X0 - 8 + 8 * h;

  Frag A;
  if (wave >= 1 && wave <= 8) {                 // interior window: b128 LDS reads
    us8 lo = *(const us8*)&s_in[m * DIM + xa];
    us8 hi = *(const us8*)&s_in[m * DIM + xa + 16];
#pragma unroll
    for (int i = 0; i < 8; ++i) { A.us[i] = lo[i]; A.us[8+i] = hi[i]; }
  } else {
#pragma unroll
    for (int i = 0; i < 8; ++i) {
      int x0 = xa + i, x1 = xa + 16 + i;
      A.us[i]   = (x0 >= 0 && x0 < DIM) ? s_in[m * DIM + x0] : (unsigned short)0;
      A.us[8+i] = (x1 >= 0 && x1 < DIM) ? s_in[m * DIM + x1] : (unsigned short)0;
    }
  }
  v16bf B;                                      // Toeplitz: B[k][n] = kern[k-n-4]
#pragma unroll
  for (int i = 0; i < 16; ++i) B[i] = (__bf16)kern1d((i < 8 ? 8*h + i : 8 + 8*h + i) - n - 4);

  v8f acc = {};
  acc = wmma_bf16(A.bf, B, acc);
#pragma unroll
  for (int r = 0; r < 8; ++r)                   // D: M=r+8h (y), N=n (x)
    s_out[(8*h + r) * DIM + X0 + n] = bf16_bits(acc[r]);
  __syncthreads();

  // cooperative coalesced writeback: one b128 per thread
  const int row = threadIdx.x / 20, seg = threadIdx.x % 20;
  us8 v = *(const us8*)&s_out[row * DIM + 8 * seg];
  *(us8*)((unsigned short*)(dst + base + (size_t)row * SY + 8 * seg)) = v;
}

// ============ pass Y: TDM-stage 160y x 16x strip, WMMA per y-tile ============
// block: (b, z, x-tile); 10 waves, wave w -> y-tile Y0=16w
__global__ void __launch_bounds__(320) ncut_conv_y(const __bf16* __restrict__ src,
                                                   __bf16* __restrict__ dst) {
  __shared__ __align__(16) unsigned short s_in[DIM * 16];   // [y][x]
  __shared__ __align__(16) unsigned short s_out[DIM * 16];
  const int blk = blockIdx.x;
  const int tx = blk % 10, z = (blk / 10) % DIM, b = blk / 1600;
  const size_t base = (size_t)b * NVOL + (size_t)z * SZ + tx * 16;

  if (threadIdx.x < 32) {
    tdm_load_2d(lds_addr_of(s_in), src + base, /*2B*/1, 16, DIM, SY);
    __builtin_amdgcn_s_wait_tensorcnt(0);
  }
  __syncthreads();

  const int wave = threadIdx.x >> 5, lane = threadIdx.x & 31;
  const int m = lane & 15, h = lane >> 4, n = m;
  const int Y0 = wave * 16;

  v16bf A;                                      // Toeplitz: A[m][k] = kern[k-m-4]
#pragma unroll
  for (int i = 0; i < 8; ++i) {
    A[i]   = (__bf16)kern1d(8*h + i      - m - 4);
    A[8+i] = (__bf16)kern1d(8*h + 16 + i - m - 4);
  }
  Frag Bf;                                      // B[k][n] = strip[Y0-8+k][n]
#pragma unroll
  for (int i = 0; i < 16; ++i) {
    int y = Y0 - 8 + (i < 8 ? 8*h + i : 8 + 8*h + i);
    Bf.us[i] = (y >= 0 && y < DIM) ? s_in[y * 16 + n] : (unsigned short)0;
  }
  v8f acc = {};
  acc = wmma_bf16(A, Bf.bf, acc);
#pragma unroll
  for (int r = 0; r < 8; ++r)                   // D: M=r+8h (y), N=n (x)
    s_out[(Y0 + 8*h + r) * 16 + n] = bf16_bits(acc[r]);
  __syncthreads();

  const int row = threadIdx.x >> 1, half = threadIdx.x & 1;
  us8 v = *(const us8*)&s_out[row * 16 + 8 * half];
  *(us8*)((unsigned short*)(dst + base + (size_t)row * SY + 8 * half)) = v;
}

// ==== pass Z: TDM-stage 160z x 16x strips (data + labels), fused reduce ====
// block: (b, y, x-tile); 10 waves, wave w -> z-tile Z0=16w
__global__ void __launch_bounds__(320) ncut_conv_z_red(const __bf16* __restrict__ src,
                                                       const float* __restrict__ labels,
                                                       float* __restrict__ zpart, int cls) {
  __shared__ __align__(16) unsigned short s_in[DIM * 16];   // [z][x] bf16
  __shared__ __align__(16) float s_lab[DIM * 16];           // [z][x] f32
  __shared__ float s_red[10];
  const int blk = blockIdx.x;
  const int tx = blk % 10, y = (blk / 10) % DIM, b = blk / 1600;
  const size_t base = (size_t)b * NVOL + (size_t)y * SY + tx * 16;

  if (threadIdx.x < 32) {
    tdm_load_2d(lds_addr_of(s_in),  src + base,    /*2B*/1, 16, DIM, SZ);
    tdm_load_2d(lds_addr_of(s_lab), labels + base, /*4B*/2, 16, DIM, SZ);
    __builtin_amdgcn_s_wait_tensorcnt(0);
  }
  __syncthreads();

  const int wave = threadIdx.x >> 5, lane = threadIdx.x & 31;
  const int m = lane & 15, h = lane >> 4, n = m;
  const int Z0 = wave * 16;

  v16bf A;
#pragma unroll
  for (int i = 0; i < 8; ++i) {
    A[i]   = (__bf16)kern1d(8*h + i      - m - 4);
    A[8+i] = (__bf16)kern1d(8*h + 16 + i - m - 4);
  }
  Frag Bf;
#pragma unroll
  for (int i = 0; i < 16; ++i) {
    int z = Z0 - 8 + (i < 8 ? 8*h + i : 8 + 8*h + i);
    Bf.us[i] = (z >= 0 && z < DIM) ? s_in[z * 16 + n] : (unsigned short)0;
  }
  v8f acc = {};
  acc = wmma_bf16(A, Bf.bf, acc);

  float s = 0.f;                                // multiply by class_probs, reduce tile
#pragma unroll
  for (int r = 0; r < 8; ++r) {
    float L = s_lab[(Z0 + 8*h + r) * 16 + n];
    float p = cls ? (1.0f - L) : L;
    s += p * acc[r];
  }
  for (int off = 16; off > 0; off >>= 1) s += __shfl_down(s, off, 32);
  if (lane == 0) s_red[wave] = s;
  __syncthreads();
  if (threadIdx.x == 0) {
    float t = 0.f;
#pragma unroll
    for (int w = 0; w < 10; ++w) t += s_red[w];
    zpart[blk] = t;
  }
}

// ============ sum block partials (deterministic tree) ============
__global__ void ncut_sumzp(const float* __restrict__ zpart,
                           float* __restrict__ scalars, int idx) {
  __shared__ float sm[256];
  const int tid = threadIdx.x;
  float v = 0.f;
  for (int i = tid; i < CBLOCKS; i += 256) v += zpart[i];
  sm[tid] = v;
  __syncthreads();
  for (int s = 128; s > 0; s >>= 1) {
    if (tid < s) sm[tid] += sm[tid+s];
    __syncthreads();
  }
  if (tid == 0) scalars[idx] = sm[0];
}

// ============ finalize ============
__global__ void ncut_final(const float* __restrict__ sc, float* __restrict__ out) {
  if (threadIdx.x == 0 && blockIdx.x == 0) {
    float l0 = fabsf(sc[0] / (sc[1] + 1e-6f));
    float l1 = fabsf(sc[2] / (sc[3] + 1e-6f));
    out[0] = 2.0f - (l0 + l1);
  }
}

extern "C" void kernel_launch(void* const* d_in, const int* in_sizes, int n_in,
                              void* d_out, int out_size, void* d_ws, size_t ws_size,
                              hipStream_t stream) {
  const float* labels = (const float*)d_in[0];
  const float* inputs = (const float*)d_in[1];

  // workspace (~67 MB): partials | means | scalars | zpart | bf16 buf0 | bf16 buf1
  char* ws = (char*)d_ws;
  float*  part    = (float*)(ws + 0);
  float*  means   = (float*)(ws + 16384);
  float*  scalars = (float*)(ws + 16640);
  float*  zpart   = (float*)(ws + 32768);
  __bf16* buf0    = (__bf16*)(ws + (size_t)(1u << 20));
  __bf16* buf1    = (__bf16*)(ws + (size_t)(1u << 20) + (size_t)(32u << 20));

  ncut_red1<<<NCHUNK, 256, 0, stream>>>(labels, inputs, part);
  ncut_red2<<<NB, 256, 0, stream>>>(part, means);

  for (int cls = 0; cls < 2; ++cls) {
    for (int q = 0; q < 2; ++q) {   // q=0: numerator field p*w, q=1: denominator field w
      ncut_weights<<<2048, 256, 0, stream>>>(labels, inputs, means, buf0, cls, q);
      ncut_conv_x<<<CBLOCKS, 320, 0, stream>>>(buf0, buf1);
      ncut_conv_y<<<CBLOCKS, 320, 0, stream>>>(buf1, buf0);
      ncut_conv_z_red<<<CBLOCKS, 320, 0, stream>>>(buf0, labels, zpart, cls);
      ncut_sumzp<<<1, 256, 0, stream>>>(zpart, scalars, cls * 2 + q);
    }
  }
  ncut_final<<<1, 64, 0, stream>>>(scalars, (float*)d_out);
}
